// LightweightSelfAttention_6476810682720
// MI455X (gfx1250) — matrix operand order repaired and checked by reading, not compiled, not run on hardware
//
#include <hip/hip_runtime.h>
#include <hip/hip_bf16.h>

// ---------------------------------------------------------------------------
// LightweightSelfAttention for MI455X (gfx1250, wave32, WMMA + TDM)
// B=8, C=256, N=1024 tokens, 4 heads, head_dim=64.
// GEMMs: v_wmma_f32_16x16x32_f16. BN/dw folded into QKV weights.
// Activations kept token-major (N x K) so every WMMA B-fragment is one
// contiguous 32B load per lane; A weight tiles staged to LDS via the
// Tensor Data Mover (tensor_load_to_lds + s_wait_tensorcnt).
// ---------------------------------------------------------------------------

typedef __attribute__((ext_vector_type(16))) _Float16 v16h;
typedef __attribute__((ext_vector_type(8)))  _Float16 v8h;
typedef __attribute__((ext_vector_type(8)))  float    v8f;
typedef __attribute__((ext_vector_type(4)))  _Float16 v4h;
typedef unsigned int u32x4 __attribute__((ext_vector_type(4)));
typedef int          i32x4 __attribute__((ext_vector_type(4)));
typedef int          i32x8 __attribute__((ext_vector_type(8)));

union frag16 { v16h v; v8h h[2]; };

#define BATCH   8
#define CH      256
#define NTOK    1024
#define NHEAD   4
#define HDIM    64
#define QKV_CH  768
#define HSTRIDE (3 * HDIM * NTOK)   // per (b,head) block in qkvT: Qt | Kt | V

__device__ inline v8f wmma_f16(v16h a, v16h b, v8f c) {
  return __builtin_amdgcn_wmma_f32_16x16x32_f16(
      false, a, false, b, (short)0, c, false, false);
}

// --- Fragment loaders (CDNA5 ISA 7.12.2) -----------------------------------
// A: 16x32 MxK from row-major MxK source. Two contiguous 16B chunks per lane.
__device__ inline v16h load_frag_a(const _Float16* __restrict__ p, int ld,
                                   int mBase, int lane) {
  const _Float16* row = p + (size_t)(mBase + (lane & 15)) * ld;
  int kh = (lane >> 4) * 16;
  frag16 u;
  u.h[0] = *(const v8h*)(row + kh);        // K kh..kh+7
  u.h[1] = *(const v8h*)(row + kh + 8);    // K kh+8..kh+15
  return u.v;
}

// B: 32x16 KxN, source stored N x K row-major -> one contiguous 32B load/lane.
__device__ inline v16h load_frag_b_t(const _Float16* __restrict__ p, int ld,
                                     int nBase, int lane) {
  int n  = nBase + (lane & 15);
  int k0 = (lane >> 4) * 16;
  return *(const v16h*)(p + (size_t)n * ld + k0);
}

// --- TDM descriptor helper: 2D tile of 16-bit data, global -> LDS ----------
// tile_dim0 halves per row, tile_dim1 rows, row stride tStride (halves),
// LDS rows padded by 4 DWORDs every padIntCode-coded interval.
__device__ inline void tdm_load_2d(const _Float16* gsrc, unsigned ldsByteOff,
                                   int tile0, int tile1, int tStride,
                                   unsigned padIntCode) {
#if __has_builtin(__builtin_amdgcn_tensor_load_to_lds)
  unsigned long long ga = (unsigned long long)(size_t)gsrc;
  u32x4 g0;
  g0.x = 1u;                                   // count=1, user descriptor
  g0.y = ldsByteOff;                           // lds_addr
  g0.z = (unsigned)(ga & 0xFFFFFFFFu);         // global_addr[31:0]
  g0.w = (unsigned)((ga >> 32) & 0x01FFFFFFu) | (2u << 30);  // addr hi | type=2
  i32x8 g1;
  g1[0] = (int)((1u << 16) | (1u << 20) | (padIntCode << 22) | (3u << 25));
  g1[1] = (int)(((unsigned)tStride & 0xFFFFu) << 16);   // tensor_dim0 lo16
  g1[2] = (int)((((unsigned)tStride >> 16) & 0xFFFFu) |
                (((unsigned)tile1 & 0xFFFFu) << 16));   // dim0 hi | dim1 lo
  g1[3] = (int)(((unsigned)tile0 & 0xFFFFu) << 16);     // dim1 hi=0 | tile_dim0
  g1[4] = (int)((unsigned)tile1 & 0xFFFFu);             // tile_dim1 | tile_dim2=0
  g1[5] = (int)(unsigned)tStride;                       // tensor_dim0_stride lo
  g1[6] = 0;
  g1[7] = 0;
  i32x4 z4 = {0, 0, 0, 0};
#if __clang_major__ >= 23
  i32x8 z8 = {0, 0, 0, 0, 0, 0, 0, 0};
  __builtin_amdgcn_tensor_load_to_lds(g0, g1, z4, z4, z8, 0);
#else
  __builtin_amdgcn_tensor_load_to_lds(g0, g1, z4, z4, 0);
#endif
#endif
}

// ---------------------------------------------------------------------------
// Kernel 1: fold BN+dw into QKV weight (f16) + bias; cast out_w to f16.
// ---------------------------------------------------------------------------
__global__ __launch_bounds__(256) void
prep_kernel(const float* __restrict__ gamma, const float* __restrict__ beta,
            const float* __restrict__ mean,  const float* __restrict__ var,
            const float* __restrict__ dw,    const float* __restrict__ pw,
            const float* __restrict__ outw,
            _Float16* __restrict__ Wq, float* __restrict__ biasQ,
            _Float16* __restrict__ Wo) {
  __shared__ float red[CH];
  int o = blockIdx.x, c = threadIdx.x;
  if (o < QKV_CH) {
    float sc = gamma[c] * rsqrtf(var[c] + 1e-5f) * dw[c];
    float sh = beta[c] * dw[c] - mean[c] * sc;
    float w  = pw[o * CH + c];
    Wq[o * CH + c] = (_Float16)(w * sc);
    red[c] = w * sh;
    __syncthreads();
#pragma unroll
    for (int s = CH / 2; s > 0; s >>= 1) {
      if (c < s) red[c] += red[c + s];
      __syncthreads();
    }
    if (c == 0) biasQ[o] = red[0];
  } else {
    int oo = o - QKV_CH;
    Wo[oo * CH + c] = (_Float16)outw[oo * CH + c];
  }
}

// ---------------------------------------------------------------------------
// Kernel 2: cast + transpose input (b,c,s) f32 -> XhT (b,s,c) f16.
// ---------------------------------------------------------------------------
__global__ __launch_bounds__(256) void
castT_kernel(const float* __restrict__ x, _Float16* __restrict__ y) {
  int i  = blockIdx.x * blockDim.x + threadIdx.x;   // (b, s, c/4)
  int c4 = (i & 63) * 4;
  int s  = (i >> 6) & (NTOK - 1);
  int b  = i >> 16;
  const float* src = x + (size_t)b * CH * NTOK + s;
  v4h h;
#pragma unroll
  for (int j = 0; j < 4; ++j) h[j] = (_Float16)src[(size_t)(c4 + j) * NTOK];
  *(v4h*)(y + ((size_t)b * NTOK + s) * CH + c4) = h;
}

// ---------------------------------------------------------------------------
// Kernel 3/5: batched GEMM  out[b] = W (MxK) x XT[b]^T  (XT stored N x K).
// 4 waves, 128x128 tile; wave 64x64 = 4x4 WMMA. A tile staged to LDS by TDM;
// B fragments read straight from global (contiguous 32B/lane, L2-resident).
// mode 0: f32 out + bias + residual.  mode 1: split-transposed QKV epilogue.
// ---------------------------------------------------------------------------
#define LDA 40   // 32 + 8 halves pad (pad_amount = 4 DWORDs)

__global__ __launch_bounds__(128, 1) void
gemm_wmma(const _Float16* __restrict__ W, const _Float16* __restrict__ XT,
          const float* __restrict__ bias, const float* __restrict__ residual,
          float* __restrict__ outF, _Float16* __restrict__ qkvT,
          int M, int N, int K, int mode) {
  __shared__ _Float16 As[128 * LDA];

  const int tid  = threadIdx.x;
  const int lane = tid & 31;
  const int wid  = tid >> 5;
  const int waveM = (wid >> 1) * 64;
  const int waveN = (wid & 1) * 64;
  const int nBase = blockIdx.x * 128;
  const int mBase = blockIdx.y * 128;
  const int b     = blockIdx.z;
  const _Float16* XTb = XT + (size_t)b * N * K;

  v8f acc[4][4] = {};

  for (int kb = 0; kb < K; kb += 32) {
    __syncthreads();
#if __has_builtin(__builtin_amdgcn_tensor_load_to_lds)
    if (wid == 0) {   // TDM issues once per wave (EXEC ignored)
      tdm_load_2d(W + (size_t)mBase * K + kb, (unsigned)(size_t)(void*)As,
                  /*tile0=*/32, /*tile1=*/128, /*tStride=*/K,
                  /*padIntCode: 16 DW interval=*/3u);
      __builtin_amdgcn_s_wait_tensorcnt(0);
    }
#else
    {   // fallback: manual staging, 64B contiguous per thread
      const _Float16* ga = W + (size_t)(mBase + tid) * K + kb;
      _Float16* la = As + tid * LDA;
#pragma unroll
      for (int i = 0; i < 4; ++i) ((float4*)la)[i] = ((const float4*)ga)[i];
    }
#endif
    if (kb + 32 < K)
      __builtin_prefetch(XTb + (size_t)(nBase + tid) * K + kb + 32, 0, 1);
    __syncthreads();

    v16h af[4], bf[4];
#pragma unroll
    for (int mi = 0; mi < 4; ++mi)
      af[mi] = load_frag_a(As, LDA, waveM + mi * 16, lane);
#pragma unroll
    for (int ni = 0; ni < 4; ++ni)
      bf[ni] = load_frag_b_t(XTb + kb, K, nBase + waveN + ni * 16, lane);
#pragma unroll
    for (int mi = 0; mi < 4; ++mi)
#pragma unroll
      for (int ni = 0; ni < 4; ++ni)
        acc[mi][ni] = wmma_f16(af[mi], bf[ni], acc[mi][ni]);
  }

  // Epilogue. C layout: M = g + 8*(lane>>4), N = lane&15.
  const int rHi = (lane >> 4) * 8;
  const int nLn = lane & 15;
#pragma unroll
  for (int mi = 0; mi < 4; ++mi) {
    const int m0 = mBase + waveM + mi * 16 + rHi;   // 8 consecutive rows m0+g
#pragma unroll
    for (int ni = 0; ni < 4; ++ni) {
      const int gn = nBase + waveN + ni * 16 + nLn;
      if (mode == 0) {          // f32 out + bias + residual
#pragma unroll
        for (int g = 0; g < 8; ++g) {
          int gm = m0 + g;
          float v = acc[mi][ni][g] + bias[gm];
          size_t idx = (size_t)b * M * N + (size_t)gm * N + gn;
          outF[idx] = v + residual[idx];
        }
      } else {                  // QKV: q,k transposed (s,d); v kept (d,s)
        int head = m0 / 192, j0 = m0 % 192;   // same region for all 8 g's
        _Float16* hb = qkvT + ((size_t)b * NHEAD + head) * HSTRIDE;
        if (j0 < 2 * HDIM) {    // Qt or Kt: 8 contiguous halves -> 16B store
          _Float16* dst = hb + (j0 < HDIM ? 0 : (HDIM * NTOK) - HDIM)
                             + (size_t)gn * HDIM + j0;
          v8h h;
#pragma unroll
          for (int g = 0; g < 8; ++g) h[g] = (_Float16)(acc[mi][ni][g] + bias[m0 + g]);
          *(v8h*)dst = h;
        } else {                // V: (d, s) rows, scattered b16 stores
          _Float16* dst = hb + 2 * HDIM * NTOK + (size_t)(j0 - 2 * HDIM) * NTOK + gn;
#pragma unroll
          for (int g = 0; g < 8; ++g)
            dst[(size_t)g * NTOK] = (_Float16)(acc[mi][ni][g] + bias[m0 + g]);
        }
      }
    }
  }
}

// ---------------------------------------------------------------------------
// Kernel 4: attention, one wave per 16-query block per (b,head).
// Qt,Kt stored (s,d); V stored (d,s) -> all fragments are vector loads.
// ---------------------------------------------------------------------------
#define SLD (NTOK + 16)

__global__ __launch_bounds__(32, 1) void
attn_kernel(const _Float16* __restrict__ qkvT, _Float16* __restrict__ attOT) {
  __shared__ _Float16 Sbuf[16 * SLD];

  const int lane  = threadIdx.x;
  const int bid   = blockIdx.x;
  const int sTile = bid & 63;
  const int head  = (bid >> 6) & (NHEAD - 1);
  const int b     = bid >> 8;
  const int sBase = sTile * 16;

  const _Float16* Qt = qkvT + ((size_t)b * NHEAD + head) * HSTRIDE;
  const _Float16* Kt = Qt + HDIM * NTOK;
  const _Float16* V  = Qt + 2 * HDIM * NTOK;

  // Pass 1: S = (Q K^T) * 0.125. A-fragments of Qt loaded once.
  v16h aQ[2];
#pragma unroll
  for (int ks = 0; ks < 2; ++ks)
    aQ[ks] = load_frag_a(Qt + ks * 32, HDIM, sBase, lane);

  const int rHi = (lane >> 4) * 8;
  const int nLn = lane & 15;
  for (int tt = 0; tt < 64; ++tt) {
    const int tBase = tt * 16;
    v8f s = {};
#pragma unroll
    for (int ks = 0; ks < 2; ++ks)
      s = wmma_f16(aQ[ks], load_frag_b_t(Kt + ks * 32, HDIM, tBase, lane), s);
#pragma unroll
    for (int g = 0; g < 8; ++g)
      Sbuf[(g + rHi) * SLD + tBase + nLn] = (_Float16)(s[g] * 0.125f);
  }
  __syncthreads();

  // Pass 2: softmax; two lanes per row, normalized in place.
  {
    const int r = lane >> 1, h = lane & 1;
    _Float16* row = Sbuf + r * SLD + h * 512;
    float mx = -3.0e38f;
    for (int t = 0; t < 512; ++t) mx = fmaxf(mx, (float)row[t]);
    mx = fmaxf(mx, __shfl_xor(mx, 1, 32));
    float sum = 0.0f;
    for (int t = 0; t < 512; ++t) {
      float e = __expf((float)row[t] - mx);
      sum += e;
      row[t] = (_Float16)e;
    }
    sum += __shfl_xor(sum, 1, 32);
    float inv = 1.0f / sum;
    for (int t = 0; t < 512; ++t) row[t] = (_Float16)((float)row[t] * inv);
  }
  __syncthreads();

  // Pass 3: O = P V^T (V rows are t-contiguous -> vector B fragments).
  v8f o[4] = {};
  for (int kt = 0; kt < 32; ++kt) {
    v16h aP = load_frag_a(Sbuf + kt * 32, SLD, 0, lane);
#pragma unroll
    for (int dt = 0; dt < 4; ++dt)
      o[dt] = wmma_f16(aP, load_frag_b_t(V + kt * 32, NTOK, dt * 16, lane), o[dt]);
  }

  // Write attOT (b, s, c) f16 (token-major for the next GEMM's B fragments).
#pragma unroll
  for (int dt = 0; dt < 4; ++dt) {
    int cch = head * HDIM + dt * 16 + nLn;
    _Float16* dst = attOT + ((size_t)b * NTOK + sBase + rHi) * CH + cch;
#pragma unroll
    for (int g = 0; g < 8; ++g) dst[(size_t)g * CH] = (_Float16)o[dt][g];
  }
}

// ---------------------------------------------------------------------------
// Launch.
// ---------------------------------------------------------------------------
extern "C" void kernel_launch(void* const* d_in, const int* in_sizes, int n_in,
                              void* d_out, int out_size, void* d_ws, size_t ws_size,
                              hipStream_t stream) {
  const float* input = (const float*)d_in[0];
  const float* gamma = (const float*)d_in[1];
  const float* beta  = (const float*)d_in[2];
  const float* mean  = (const float*)d_in[3];
  const float* var   = (const float*)d_in[4];
  const float* dw    = (const float*)d_in[5];
  const float* pw    = (const float*)d_in[6];
  const float* outw  = (const float*)d_in[7];
  const float* outb  = (const float*)d_in[8];

  char* ws = (char*)d_ws;
  size_t off = 0;
  auto alloc = [&](size_t bytes) {
    void* p = ws + off;
    off = (off + bytes + 255) & ~(size_t)255;
    return p;
  };
  _Float16* Wq    = (_Float16*)alloc((size_t)QKV_CH * CH * 2);
  _Float16* Wo    = (_Float16*)alloc((size_t)CH * CH * 2);
  float*    biasQ = (float*)   alloc((size_t)QKV_CH * 4);
  _Float16* XhT   = (_Float16*)alloc((size_t)BATCH * NTOK * CH * 2);
  _Float16* qkvT  = (_Float16*)alloc((size_t)BATCH * NHEAD * HSTRIDE * 2);
  _Float16* attOT = (_Float16*)alloc((size_t)BATCH * NTOK * CH * 2);

  prep_kernel<<<QKV_CH + CH, CH, 0, stream>>>(gamma, beta, mean, var, dw, pw,
                                              outw, Wq, biasQ, Wo);
  castT_kernel<<<(BATCH * NTOK * 64) / 256, 256, 0, stream>>>(input, XhT);

  // QKV projection -> split-transposed q/k + row-major v
  gemm_wmma<<<dim3(NTOK / 128, QKV_CH / 128, BATCH), 128, 0, stream>>>(
      Wq, XhT, biasQ, nullptr, nullptr, qkvT, QKV_CH, NTOK, CH, 1);

  attn_kernel<<<BATCH * NHEAD * (NTOK / 16), 32, 0, stream>>>(qkvT, attOT);

  // Output projection + bias + residual -> f32
  gemm_wmma<<<dim3(NTOK / 128, CH / 128, BATCH), 128, 0, stream>>>(
      Wo, attOT, outb, input, (float*)d_out, nullptr, CH, NTOK, CH, 0);
}
